// MyModel_15212774163235
// MI455X (gfx1250) — matrix operand order repaired and checked by reading, not compiled, not run on hardware
//
#include <hip/hip_runtime.h>

// ---------------------------------------------------------------------------
// MI455X (gfx1250) implementation:
//   h1 = sine(x @ W1^T + b1)          bf16 WMMA GEMM, fused sine epilogue
//   h2 = sine(h1 @ W2^T + b2)         bf16 WMMA GEMM, fused sine + Bernoulli
//   xb = (noise < h2) ? 1 : 0
//   out[b] = xb_b^T Q xb_b            bf16 WMMA GEMM vs Q^T, fused dot+reduce
// V_WMMA_F32_16X16X32_BF16 + double-buffered LDS fed by async global->LDS DMA.
// ---------------------------------------------------------------------------

typedef __attribute__((ext_vector_type(16))) __bf16 v16bf;
typedef __attribute__((ext_vector_type(8)))  float  v8f;
typedef __attribute__((ext_vector_type(4)))  int    v4i;

union Frag {
    v16bf v;
    uint4 q[2];
};

__device__ __forceinline__ float bf16_to_f32(unsigned short h) {
    return __uint_as_float(((unsigned int)h) << 16);
}
__device__ __forceinline__ unsigned short f32_to_bf16(float f) {
    unsigned int u = __float_as_uint(f);
    u += 0x7FFFu + ((u >> 16) & 1u);       // round-to-nearest-even
    return (unsigned short)(u >> 16);
}

#define PI_F 3.14159265358979323846f

// ---- CDNA5 async global->LDS copy (no VGPR staging, tracked by ASYNCcnt) ---
#if defined(__gfx1250__) && __has_builtin(__builtin_amdgcn_global_load_async_to_lds_b128)
#define HAVE_ASYNC_LDS 1
#else
#define HAVE_ASYNC_LDS 0
#endif

#if HAVE_ASYNC_LDS
typedef __attribute__((address_space(1))) v4i* g128p;   // global 128-bit chunk
typedef __attribute__((address_space(3))) v4i* l128p;   // LDS 128-bit chunk
#endif

__device__ __forceinline__ void copy16_g2l(const unsigned short* g,
                                           unsigned short* l) {
#if HAVE_ASYNC_LDS
    __builtin_amdgcn_global_load_async_to_lds_b128((g128p)g, (l128p)l, 0, 0);
#else
    *(uint4*)l = *(const uint4*)g;
#endif
}

__device__ __forceinline__ void wait_async_all() {
#if HAVE_ASYNC_LDS
#if __has_builtin(__builtin_amdgcn_s_wait_asynccnt)
    __builtin_amdgcn_s_wait_asynccnt(0);
#else
    asm volatile("s_wait_asynccnt 0x0" ::: "memory");
#endif
#endif
}

// --------------------------- helper kernels ---------------------------------

__global__ void cvt_f32_bf16_kernel(const float* __restrict__ src,
                                    unsigned short* __restrict__ dst, int count) {
    int i = blockIdx.x * blockDim.x + threadIdx.x;
    int stride = gridDim.x * blockDim.x;
    for (; i < count; i += stride) dst[i] = f32_to_bf16(src[i]);
}

// dst[j*dim + i] = bf16(src[i*dim + j])
__global__ void transpose_cvt_kernel(const float* __restrict__ src,
                                     unsigned short* __restrict__ dst, int dim) {
    __shared__ float tile[32][33];
    const int bx = blockIdx.x * 32, by = blockIdx.y * 32;
    const int tx = threadIdx.x, ty = threadIdx.y;   // block = (32, 8)
    #pragma unroll
    for (int i = 0; i < 32; i += 8)
        tile[ty + i][tx] = src[(size_t)(by + ty + i) * dim + bx + tx];
    __syncthreads();
    #pragma unroll
    for (int i = 0; i < 32; i += 8)
        dst[(size_t)(bx + ty + i) * dim + by + tx] = f32_to_bf16(tile[tx][ty + i]);
}

// out[r] = sum of `parts` partials (deterministic final reduce)
__global__ void reduce_out_kernel(const float* __restrict__ P,
                                  float* __restrict__ out, int rows, int parts) {
    int r = blockIdx.x * blockDim.x + threadIdx.x;
    if (r < rows) {
        float s = 0.0f;
        for (int i = 0; i < parts; ++i) s += P[(size_t)r * parts + i];
        out[r] = s;
    }
}

// ------------------------------- GEMM kernel --------------------------------
// C[m,n] = sum_k A[m,k] * B[n,k]   (A: [M x K] bf16, B: [N x K] bf16)
// MODE 0: Out16 = bf16(sine(C + bias[n]))
// MODE 1: Out16 = bf16((noise[m,n] < sine(C + bias[n])) ? 1 : 0)
// MODE 2: Pout[m*4*gridDim.x + blockIdx.x*4 + wn] = sum_n C[m,n]*xb[m,n]
//
// Block tile 128x128, 8 waves as 2(M) x 4(N), wave tile 64x32 (4x2 accums).
// K staged 32 deep into double-buffered LDS; prefetch overlaps compute.

constexpr int BM = 128, BN = 128, BK = 32, LDT = 40;   // LDT: padded LDS row (bf16)

template <int MODE>
__global__ __launch_bounds__(256) void gemm_bf16_kernel(
    const unsigned short* __restrict__ A,      // [M x K]
    const unsigned short* __restrict__ B,      // [N x K]
    const float* __restrict__ bias,            // [N]       (MODE 0/1)
    const float* __restrict__ noise,           // [M x N]   (MODE 1)
    const unsigned short* __restrict__ XBm,    // [M x N]   (MODE 2)
    unsigned short* __restrict__ Out16,        // [M x N]   (MODE 0/1)
    float* __restrict__ Pout,                  // [M x 4*gridN] (MODE 2)
    int M, int N, int K)
{
    __shared__ unsigned short As[2 * BM * LDT];   // 20 KB (double-buffered)
    __shared__ unsigned short Bs[2 * BN * LDT];   // 20 KB

    const int tid  = threadIdx.x;
    const int lane = tid & 31;             // wave32
    const int wid  = tid >> 5;             // 8 waves
    const int wm   = wid & 1;              // 2 waves along M (64 rows each)
    const int wn   = wid >> 1;             // 4 waves along N (32 cols each)
    const int half = lane >> 4;            // 0 / 1 (K-half split per ISA layout)
    const int l16  = lane & 15;

    const int m0 = blockIdx.y * BM;
    const int n0 = blockIdx.x * BN;

    v8f zacc = {};
    v8f acc[4][2];
    #pragma unroll
    for (int i = 0; i < 4; ++i)
        #pragma unroll
        for (int j = 0; j < 2; ++j) acc[i][j] = zacc;

    // global -> LDS loaders: 2 threads/row, 16 bf16 (32B) each, for A and B
    const int arow = tid >> 1;             // 0..127
    const int aseg = (tid & 1) * 16;       // 0 / 16
    const unsigned short* Ag = A + (size_t)(m0 + arow) * K + aseg;
    const unsigned short* Bg = B + (size_t)(n0 + arow) * K + aseg;
    unsigned short* AsW = &As[arow * LDT + aseg];
    unsigned short* BsW = &Bs[arow * LDT + aseg];

    const int nsteps = K / BK;

    // prologue: stage tile 0 into buffer 0
    copy16_g2l(Ag, AsW);
    copy16_g2l(Ag + 8, AsW + 8);
    copy16_g2l(Bg, BsW);
    copy16_g2l(Bg + 8, BsW + 8);

    for (int t = 0; t < nsteps; ++t) {
        const int buf  = t & 1;
        const int bufA = buf * BM * LDT;
        const int bufB = buf * BN * LDT;

        wait_async_all();                  // my async writes for tile t done
        __syncthreads();                   // tile t visible; prev reads retired

        if (t + 1 < nsteps) {              // prefetch tile t+1 into other buffer
            const int kk  = (t + 1) * BK;
            const int obA = (1 - buf) * BM * LDT;
            const int obB = (1 - buf) * BN * LDT;
            copy16_g2l(Ag + kk,     AsW + obA);
            copy16_g2l(Ag + kk + 8, AsW + obA + 8);
            copy16_g2l(Bg + kk,     BsW + obB);
            copy16_g2l(Bg + kk + 8, BsW + obB + 8);
        }

        // A fragments (16x32): lanes 0-15 hold K 0-7 & 16-23, lanes 16-31 K 8-15 & 24-31
        Frag af[4];
        #pragma unroll
        for (int mt = 0; mt < 4; ++mt) {
            int base = bufA + (wm * 64 + mt * 16 + l16) * LDT;
            af[mt].q[0] = *(const uint4*)&As[base + half * 8];
            af[mt].q[1] = *(const uint4*)&As[base + 16 + half * 8];
        }
        // B fragments (32x16): lanes 0-15 hold K 0-7 & 16-23, lanes 16-31 K 8-15 & 24-31 (col = l16)
        Frag bf[2];
        #pragma unroll
        for (int nt = 0; nt < 2; ++nt) {
            int base = bufB + (wn * 32 + nt * 16 + l16) * LDT;
            bf[nt].q[0] = *(const uint4*)&Bs[base + half * 8];
            bf[nt].q[1] = *(const uint4*)&Bs[base + 16 + half * 8];
        }
        #pragma unroll
        for (int mt = 0; mt < 4; ++mt)
            #pragma unroll
            for (int nt = 0; nt < 2; ++nt)
                acc[mt][nt] = __builtin_amdgcn_wmma_f32_16x16x32_bf16(
                    false, af[mt].v, false, bf[nt].v,
                    (short)0, acc[mt][nt], false, false);
    }

    // C/D layout: element r of v8f -> m = base + r + (lane>=16 ? 8 : 0), n = base + (lane&15)
    if (MODE == 0 || MODE == 1) {
        #pragma unroll
        for (int mt = 0; mt < 4; ++mt) {
            #pragma unroll
            for (int nt = 0; nt < 2; ++nt) {
                const int n  = n0 + wn * 32 + nt * 16 + l16;
                const float bv = bias[n];
                #pragma unroll
                for (int r = 0; r < 8; ++r) {
                    const int m = m0 + wm * 64 + mt * 16 + r + half * 8;
                    float v = acc[mt][nt][r] + bv;
                    float h = 0.5f * (1.0f + __sinf((v - 0.5f) * PI_F));
                    if (MODE == 1)
                        h = (noise[(size_t)m * N + n] < h) ? 1.0f : 0.0f;
                    Out16[(size_t)m * N + n] = f32_to_bf16(h);
                }
            }
        }
    } else {
        const int parts = 4 * gridDim.x;
        const int slot  = blockIdx.x * 4 + wn;
        #pragma unroll
        for (int mt = 0; mt < 4; ++mt) {
            float rs[8];
            #pragma unroll
            for (int r = 0; r < 8; ++r) rs[r] = 0.0f;
            #pragma unroll
            for (int nt = 0; nt < 2; ++nt) {
                const int n = n0 + wn * 32 + nt * 16 + l16;
                #pragma unroll
                for (int r = 0; r < 8; ++r) {
                    const int m = m0 + wm * 64 + mt * 16 + r + half * 8;
                    rs[r] += acc[mt][nt][r] * bf16_to_f32(XBm[(size_t)m * N + n]);
                }
            }
            #pragma unroll
            for (int r = 0; r < 8; ++r) {
                float s = rs[r];
                s += __shfl_xor(s, 1, 32);   // masks < 16 keep reduction inside
                s += __shfl_xor(s, 2, 32);   // each 16-lane N-group
                s += __shfl_xor(s, 4, 32);
                s += __shfl_xor(s, 8, 32);
                if (l16 == 0) {
                    const int m = m0 + wm * 64 + mt * 16 + r + half * 8;
                    Pout[(size_t)m * parts + slot] = s;
                }
            }
        }
    }
}

// ------------------------------ host launcher -------------------------------

extern "C" void kernel_launch(void* const* d_in, const int* in_sizes, int n_in,
                              void* d_out, int out_size, void* d_ws, size_t ws_size,
                              hipStream_t stream) {
    (void)in_sizes; (void)n_in; (void)out_size; (void)ws_size;

    constexpr int BATCH = 1024, BITS = 2048;

    const float* x     = (const float*)d_in[0];
    const float* noise = (const float*)d_in[1];
    const float* W1    = (const float*)d_in[2];
    const float* b1    = (const float*)d_in[3];
    const float* W2    = (const float*)d_in[4];
    const float* b2    = (const float*)d_in[5];
    const float* Q     = (const float*)d_in[6];
    float* out = (float*)d_out;

    // workspace layout (bf16 stored as ushort)
    unsigned short* Wb1 = (unsigned short*)d_ws;                 // 2048x2048
    unsigned short* Wb2 = Wb1 + (size_t)BITS * BITS;             // 2048x2048
    unsigned short* Qtb = Wb2 + (size_t)BITS * BITS;             // 2048x2048 (Q^T)
    unsigned short* Xb  = Qtb + (size_t)BITS * BITS;             // 1024x2048
    unsigned short* H1  = Xb  + (size_t)BATCH * BITS;            // 1024x2048
    unsigned short* XB  = H1  + (size_t)BATCH * BITS;            // 1024x2048 (binary)
    float* Pt = (float*)(XB + (size_t)BATCH * BITS);             // 1024 x 64 partials

    // ---- precision conversion (one pass; all subsequent GEMMs are bf16) ----
    cvt_f32_bf16_kernel<<<2048, 256, 0, stream>>>(W1, Wb1, BITS * BITS);
    cvt_f32_bf16_kernel<<<2048, 256, 0, stream>>>(W2, Wb2, BITS * BITS);
    cvt_f32_bf16_kernel<<<1024, 256, 0, stream>>>(x,  Xb,  BATCH * BITS);
    transpose_cvt_kernel<<<dim3(BITS / 32, BITS / 32), dim3(32, 8), 0, stream>>>(Q, Qtb, BITS);

    const dim3 ggrid(BITS / BN, BATCH / BM);   // (16, 8) = 128 blocks
    const dim3 gblk(256);

    // layer 1: H1 = sine(x @ W1^T + b1)
    gemm_bf16_kernel<0><<<ggrid, gblk, 0, stream>>>(
        Xb, Wb1, b1, nullptr, nullptr, H1, nullptr, BATCH, BITS, BITS);

    // layer 2 + Bernoulli sample: XB = (noise < sine(H1 @ W2^T + b2)) ? 1 : 0
    gemm_bf16_kernel<1><<<ggrid, gblk, 0, stream>>>(
        H1, Wb2, b2, noise, nullptr, XB, nullptr, BATCH, BITS, BITS);

    // QUBO cost: Pt[b, slot] = partial of sum_j (sum_i xb[b,i] Q[i,j]) xb[b,j]
    gemm_bf16_kernel<2><<<ggrid, gblk, 0, stream>>>(
        XB, Qtb, nullptr, nullptr, XB, nullptr, Pt, BATCH, BITS, BITS);

    // deterministic final reduce into d_out
    reduce_out_kernel<<<(BATCH + 255) / 256, 256, 0, stream>>>(Pt, out, BATCH, 4 * (BITS / BN));
}
